// LSTMDecoder_27187142983967
// MI455X (gfx1250) — compile-verified, matrix operand order checked by reading, not run on hardware
//
#include <hip/hip_runtime.h>

// ---------------------------------------------------------------------------
// LSTM decoder for MI455X (gfx1250, wave32).
// GEMMs: v_wmma_f32_16x16x32_f16 (f16 in, f32 accumulate).
// Weights converted to f16 once (28MB -> resident in 192MB L2).
// Per block (8 waves): 64(M) x 128(N) C tile. The A slab (64 x Kc) is staged
// into LDS with GLOBAL_LOAD_ASYNC_TO_LDS_B128 (ASYNCcnt / s_wait_asynccnt),
// double-buffered so the async copy of chunk k+1 overlaps WMMAs on chunk k.
// B (weights) loads are direct from L2, one full 128-deep chunk per wave so
// loads pipeline across 4 WMMAs instead of stalling each one.
// ---------------------------------------------------------------------------

typedef __attribute__((ext_vector_type(16))) _Float16 v16h;
typedef __attribute__((ext_vector_type(8)))  _Float16 v8h;
typedef __attribute__((ext_vector_type(8)))  float    v8f;

namespace cfg {
constexpr int B = 512;    // batch
constexpr int I = 1024;   // input
constexpr int H = 1024;   // hidden
constexpr int O = 2048;   // output
constexpr int T = 32;     // steps
constexpr int G = 4 * H;  // gate width 4096
}

#define KC 128  // K-chunk staged in LDS per double-buffer slot

#if defined(__HIP_DEVICE_COMPILE__) && \
    __has_builtin(__builtin_amdgcn_global_load_async_to_lds_b128)
#define USE_ASYNC_LDS 1
#else
#define USE_ASYNC_LDS 0
#endif

// Builtin signature (from hipcc diagnostic): param0 is a 16-byte int vector
// pointer in the global (AS1 / __device__) address space; param1 is the LDS
// (AS3) side; params 2,3 are immediate offset / cache-policy.
typedef int v4i __attribute__((vector_size(16)));
typedef __attribute__((address_space(1))) v4i as1_v4i;
typedef __attribute__((address_space(3))) v4i as3_v4i;

template <int N>
__device__ __forceinline__ void wait_async() {
#if USE_ASYNC_LDS
#if __has_builtin(__builtin_amdgcn_s_wait_asynccnt)
  __builtin_amdgcn_s_wait_asynccnt((unsigned short)N);
#else
  asm volatile("s_wait_asynccnt %0" ::"n"(N));
#endif
#endif
}

// ---------------------------------------------------------------------------
// f32 -> f16 conversion (one-time for weights/x)
// ---------------------------------------------------------------------------
__global__ void f32_to_f16_kernel(const float* __restrict__ src,
                                  _Float16* __restrict__ dst, int n) {
  int i = blockIdx.x * blockDim.x + threadIdx.x;
  if (i < n) dst[i] = (_Float16)src[i];
}

// ---------------------------------------------------------------------------
// WMMA GEMM: out[M,N] = A[M,K] * W[N,K]^T (+ Cinit) (+ bias[N])
// Block: 32x8 (8 waves, 256 threads). Wave w owns tileN = blockIdx.x*8 + w,
// all waves share the 64-row M slab rowM0 = blockIdx.y*64.
// ---------------------------------------------------------------------------
__global__ __launch_bounds__(256) void wmma_gemm_bias_kernel(
    const _Float16* __restrict__ A, const _Float16* __restrict__ W,
    const float* __restrict__ Cinit, const float* __restrict__ bias,
    float* __restrict__ out, int N, int K) {
  __shared__ _Float16 aT[2][64 * KC];  // 2 x 16KB double buffer

  const int lane  = threadIdx.x;              // 0..31 (wave32)
  const int m     = lane & 15;                // row/col within 16-tile
  const int halfw = lane >> 4;                // K-interleave half
  const int wave  = threadIdx.y;              // 0..7
  const int tid   = wave * 32 + lane;         // 0..255
  const int tileN = blockIdx.x * 8 + wave;
  const int rowM0 = blockIdx.y * 64;
  const int col   = tileN * 16 + m;

  // B fragment base: row n of W (= column n of W^T); CDNA5 16-bit layout:
  // lanes 0..15 K 0..7/16..23, lanes 16..31 K 8..15/24..31 -> +halfw*8.
  const _Float16* wr = W + (size_t)col * K + halfw * 8;

  v8f acc0 = {}, acc1 = {}, acc2 = {}, acc3 = {};
  if (Cinit) {
#pragma unroll
    for (int r = 0; r < 8; ++r) {
      const int mr = r + 8 * halfw;  // C/D layout: VGPR r <-> M = r + 8*half
      acc0[r] = Cinit[(size_t)(rowM0 +  0 + mr) * N + col];
      acc1[r] = Cinit[(size_t)(rowM0 + 16 + mr) * N + col];
      acc2[r] = Cinit[(size_t)(rowM0 + 32 + mr) * N + col];
      acc3[r] = Cinit[(size_t)(rowM0 + 48 + mr) * N + col];
    }
  }

  // Cooperative stage of one 64 x KC f16 A-chunk into aT[buf].
  // 16KB / 256 threads = 4 x 16B async transfers per thread.
  auto stage = [&](int buf, int kc) {
#pragma unroll
    for (int i = 0; i < 4; ++i) {
      const int e    = (tid + i * 256) * 8;  // half-index within 64xKC tile
      const int row  = e / KC;
      const int colh = e % KC;
      const _Float16* src = A + (size_t)(rowM0 + row) * K + kc + colh;
      _Float16* dst = &aT[buf][e];
#if USE_ASYNC_LDS
      __builtin_amdgcn_global_load_async_to_lds_b128(
          (as1_v4i*)(const void*)src, (as3_v4i*)(void*)dst, 0, 0);
#else
      *(v8h*)dst = *(const v8h*)src;
#endif
    }
  };

  stage(0, 0);  // prologue: chunk 0 in flight
  int buf = 0;
  for (int kc = 0; kc < K; kc += KC) {
    const bool haveNext = (kc + KC) < K;
    if (haveNext) {
      stage(buf ^ 1, kc + KC);  // overlap next chunk's copy with compute
      wait_async<4>();          // this wave's *current* chunk loads done
    } else {
      wait_async<0>();
    }
    __syncthreads();            // all waves' staged data visible

    const _Float16* as = &aT[buf][0] + halfw * 8;
#pragma unroll
    for (int kk = 0; kk < KC; kk += 32) {
      const int k0 = kc + kk;
      if (haveNext) __builtin_prefetch(wr + k0 + KC, 0, 3);  // next B chunk

      v8h blo = *(const v8h*)(wr + k0);
      v8h bhi = *(const v8h*)(wr + k0 + 16);
      v16h bv = __builtin_shufflevector(blo, bhi, 0, 1, 2, 3, 4, 5, 6, 7,
                                        8, 9, 10, 11, 12, 13, 14, 15);

      // A fragments for the 4 stacked 16x32 tiles, from LDS (ds_load_b128).
      v8h l0 = *(const v8h*)(as + (size_t)( 0 + m) * KC + kk);
      v8h h0 = *(const v8h*)(as + (size_t)( 0 + m) * KC + kk + 16);
      v8h l1 = *(const v8h*)(as + (size_t)(16 + m) * KC + kk);
      v8h h1 = *(const v8h*)(as + (size_t)(16 + m) * KC + kk + 16);
      v8h l2 = *(const v8h*)(as + (size_t)(32 + m) * KC + kk);
      v8h h2 = *(const v8h*)(as + (size_t)(32 + m) * KC + kk + 16);
      v8h l3 = *(const v8h*)(as + (size_t)(48 + m) * KC + kk);
      v8h h3 = *(const v8h*)(as + (size_t)(48 + m) * KC + kk + 16);
      v16h av0 = __builtin_shufflevector(l0, h0, 0, 1, 2, 3, 4, 5, 6, 7,
                                         8, 9, 10, 11, 12, 13, 14, 15);
      v16h av1 = __builtin_shufflevector(l1, h1, 0, 1, 2, 3, 4, 5, 6, 7,
                                         8, 9, 10, 11, 12, 13, 14, 15);
      v16h av2 = __builtin_shufflevector(l2, h2, 0, 1, 2, 3, 4, 5, 6, 7,
                                         8, 9, 10, 11, 12, 13, 14, 15);
      v16h av3 = __builtin_shufflevector(l3, h3, 0, 1, 2, 3, 4, 5, 6, 7,
                                         8, 9, 10, 11, 12, 13, 14, 15);

      acc0 = __builtin_amdgcn_wmma_f32_16x16x32_f16(false, av0, false, bv,
                                                    (short)0, acc0, false, false);
      acc1 = __builtin_amdgcn_wmma_f32_16x16x32_f16(false, av1, false, bv,
                                                    (short)0, acc1, false, false);
      acc2 = __builtin_amdgcn_wmma_f32_16x16x32_f16(false, av2, false, bv,
                                                    (short)0, acc2, false, false);
      acc3 = __builtin_amdgcn_wmma_f32_16x16x32_f16(false, av3, false, bv,
                                                    (short)0, acc3, false, false);
    }
    __syncthreads();  // everyone done reading aT[buf] before re-staging it
    buf ^= 1;
  }

  const float bcol = bias ? bias[col] : 0.0f;
#pragma unroll
  for (int r = 0; r < 8; ++r) {
    const int mr = r + 8 * halfw;
    out[(size_t)(rowM0 +  0 + mr) * N + col] = acc0[r] + bcol;
    out[(size_t)(rowM0 + 16 + mr) * N + col] = acc1[r] + bcol;
    out[(size_t)(rowM0 + 32 + mr) * N + col] = acc2[r] + bcol;
    out[(size_t)(rowM0 + 48 + mr) * N + col] = acc3[r] + bcol;
  }
}

// ---------------------------------------------------------------------------
// Fused LSTM cell pointwise update (torch gate order i,f,g,o):
//   c' = sig(f)*c + sig(i)*tanh(g);  h' = sig(o)*tanh(c')
// Also emits the f16 copy of h for the next step's WMMA GEMMs.
// ---------------------------------------------------------------------------
__global__ void lstm_cell_kernel(const float* __restrict__ gates,  // [B,4H]
                                 float* __restrict__ c,            // [B,H]
                                 float* __restrict__ h,            // [B,H]
                                 _Float16* __restrict__ h16) {     // [B,H]
  int idx = blockIdx.x * blockDim.x + threadIdx.x;  // < B*H
  int b = idx >> 10;          // /H (H=1024)
  int j = idx & (cfg::H - 1);
  const float* g = gates + (size_t)b * cfg::G;
  float gi = g[j];
  float gf = g[cfg::H + j];
  float gg = g[2 * cfg::H + j];
  float go = g[3 * cfg::H + j];
  float si = 1.0f / (1.0f + __expf(-gi));
  float sf = 1.0f / (1.0f + __expf(-gf));
  float so = 1.0f / (1.0f + __expf(-go));
  float cn = sf * c[idx] + si * tanhf(gg);
  float hn = so * tanhf(cn);
  c[idx] = cn;
  h[idx] = hn;
  h16[idx] = (_Float16)hn;
}

// ---------------------------------------------------------------------------
// kernel_launch
// ---------------------------------------------------------------------------
extern "C" void kernel_launch(void* const* d_in, const int* in_sizes, int n_in,
                              void* d_out, int out_size, void* d_ws,
                              size_t ws_size, hipStream_t stream) {
  using namespace cfg;
  (void)in_sizes; (void)n_in; (void)out_size; (void)ws_size;

  const float* x     = (const float*)d_in[0];   // [B,I]
  const float* W_ih0 = (const float*)d_in[1];   // [4H,I]
  const float* W_hh0 = (const float*)d_in[2];   // [4H,H]
  const float* b_ih0 = (const float*)d_in[3];   // [4H]
  const float* b_hh0 = (const float*)d_in[4];   // [4H]
  const float* W_ih1 = (const float*)d_in[5];   // [4H,H]
  const float* W_hh1 = (const float*)d_in[6];   // [4H,H]
  const float* b_ih1 = (const float*)d_in[7];   // [4H]
  const float* b_hh1 = (const float*)d_in[8];   // [4H]
  const float* W_out = (const float*)d_in[9];   // [O,H]
  const float* b_out = (const float*)d_in[10];  // [O]
  float* preds = (float*)d_out;                 // [T,B,O]

  // ---- workspace carve-up ----
  char* ws = (char*)d_ws;
  size_t off = 0;
  auto carve = [&](size_t bytes) { void* p = ws + off; off += bytes; return p; };

  float*    xW0   = (float*)carve((size_t)B * G * 4);   // hoisted x@W_ih0^T+b
  float*    gates = (float*)carve((size_t)B * G * 4);
  float*    h0    = (float*)carve((size_t)B * H * 4);   // state block start
  float*    c0    = (float*)carve((size_t)B * H * 4);
  float*    h1    = (float*)carve((size_t)B * H * 4);
  float*    c1    = (float*)carve((size_t)B * H * 4);
  _Float16* h0h   = (_Float16*)carve((size_t)B * H * 2);
  _Float16* h1h   = (_Float16*)carve((size_t)B * H * 2); // state block end
  _Float16* xh    = (_Float16*)carve((size_t)B * I * 2);
  _Float16* wih0h = (_Float16*)carve((size_t)G * I * 2);
  _Float16* whh0h = (_Float16*)carve((size_t)G * H * 2);
  _Float16* wih1h = (_Float16*)carve((size_t)G * H * 2);
  _Float16* whh1h = (_Float16*)carve((size_t)G * H * 2);
  _Float16* wouth = (_Float16*)carve((size_t)O * H * 2);

  // Zero h0,c0,h1,c1,h0h,h1h (contiguous) — deterministic per call.
  size_t state_bytes = (size_t)B * H * 4 * 4 + (size_t)B * H * 2 * 2;
  (void)hipMemsetAsync(h0, 0, state_bytes, stream);

  // ---- one-time conversions to f16 ----
  auto cvt = [&](const float* s, _Float16* d, int n) {
    f32_to_f16_kernel<<<(n + 255) / 256, 256, 0, stream>>>(s, d, n);
  };
  cvt(x,     xh,    B * I);
  cvt(W_ih0, wih0h, G * I);
  cvt(W_hh0, whh0h, G * H);
  cvt(W_ih1, wih1h, G * H);
  cvt(W_hh1, whh1h, G * H);
  cvt(W_out, wouth, O * H);

  // ---- GEMM launcher: block 32x8 (8 waves), block tile 64M x 128N ----
  const dim3 blk(32, 8);
  auto gemm = [&](const _Float16* A, const _Float16* W, const float* Ci,
                  const float* bias, float* out, int N, int K) {
    dim3 grd(N / 128, B / 64);
    wmma_gemm_bias_kernel<<<grd, blk, 0, stream>>>(A, W, Ci, bias, out, N, K);
  };

  // Hoisted layer-0 input projection: xW0 = x @ W_ih0^T + b_ih0
  gemm(xh, wih0h, nullptr, b_ih0, xW0, G, I);

  const int cellGrid = (B * H) / 256;
  for (int t = 0; t < T; ++t) {
    // layer 0: gates = xW0 + h0 @ W_hh0^T + b_hh0
    gemm(h0h, whh0h, xW0, b_hh0, gates, G, H);
    lstm_cell_kernel<<<cellGrid, 256, 0, stream>>>(gates, c0, h0, h0h);

    // layer 1: gates = h0 @ W_ih1^T + b_ih1 + h1 @ W_hh1^T + b_hh1
    gemm(h0h, wih1h, nullptr, b_ih1, gates, G, H);
    gemm(h1h, whh1h, gates,   b_hh1, gates, G, H);
    lstm_cell_kernel<<<cellGrid, 256, 0, stream>>>(gates, c1, h1, h1h);

    // prediction: preds[t] = h1 @ W_out^T + b_out
    gemm(h1h, wouth, nullptr, b_out, preds + (size_t)t * B * O, O, H);
  }
}
